// HashGridEncoder_58626303590455
// MI455X (gfx1250) — matrix operand order repaired and checked by reading, not compiled, not run on hardware
//
#include <hip/hip_runtime.h>

typedef _Float16 v16h __attribute__((ext_vector_type(16)));
typedef _Float16 v8h  __attribute__((ext_vector_type(8)));
typedef float    v8f  __attribute__((ext_vector_type(8)));

#define MAX_TABLE 524288u

// Geometric schedule: R_l = floor(16 * 128^(l/15)) (computed in double, matches jnp)
static constexpr unsigned RES[16]   = {16u, 22u, 30u, 42u, 58u, 80u, 111u, 153u,
                                       212u, 294u, 406u, 561u, 776u, 1072u, 1482u, 2048u};
// Levels 0..5: T = R^3 and idx < R^3 always -> mod is identity (mask all-ones).
// Levels 6..15: T = 2^19 -> mask. (2^19 | 2^32 so u32 wraparound stays exact.)
static constexpr unsigned MASKV[16] = {0xFFFFFFFFu, 0xFFFFFFFFu, 0xFFFFFFFFu, 0xFFFFFFFFu,
                                       0xFFFFFFFFu, 0xFFFFFFFFu, 0x7FFFFu, 0x7FFFFu,
                                       0x7FFFFu, 0x7FFFFu, 0x7FFFFu, 0x7FFFFu,
                                       0x7FFFFu, 0x7FFFFu, 0x7FFFFu, 0x7FFFFu};

// A-matrix f16 16x32 layout: contiguous K = 16*hi + h (matches B fills).
__device__ __forceinline__ int a_kmap(int h, int hi) {
  return h + 16 * hi;
}

// Prepack W1^T tiles (8) and W2^T tiles (16) into WMMA A-operand layout.
// ws layout: [24 tiles][32 lanes][16 halves] = 24 KB
__global__ __launch_bounds__(256) void hg_prepack_kernel(
    const float* __restrict__ W1, const float* __restrict__ W2,
    _Float16* __restrict__ pk) {
  int t = blockIdx.x * 256 + threadIdx.x;
  if (t >= 24 * 32) return;
  int tile = t >> 5, lane = t & 31;
  int hi = (lane >> 4) & 1, nloc = lane & 15;
#pragma unroll
  for (int h = 0; h < 16; ++h) {
    int K = a_kmap(h, hi);
    float v;
    if (tile < 8) {
      v = W1[K * 128 + tile * 16 + nloc];              // A1 tile n: W1[K][16n+M]
    } else {
      int ti = tile - 8;
      int n2 = ti >> 2, kc = ti & 3;
      v = W2[(kc * 32 + K) * 64 + n2 * 16 + nloc];     // A2 tile (n2,kc): W2[32kc+K'][16n2+M]
    }
    pk[t * 16 + h] = (_Float16)v;
  }
}

__global__ __launch_bounds__(256) void hg_encode_mlp_kernel(
    const float* __restrict__ coords, const float* __restrict__ tables,
    const float* __restrict__ b1, const float* __restrict__ b2,
    const _Float16* __restrict__ pk, float* __restrict__ out, int npts) {
  __shared__ alignas(16) _Float16 sw[24 * 32 * 16];   // 24 KB packed weights

  const int tid  = threadIdx.x;
  const int lane = tid & 31;
  const int wave = tid >> 5;
  const int hi   = (lane >> 4) & 1;   // high half-wave flag
  const int nloc = lane & 15;         // point-in-tile / N index
  const int off  = hi << 3;           // C/D row offset for this half-wave

  // ---- stage packed weights to LDS via CDNA5 async copy (ASYNCcnt path) ----
  {
    unsigned ldsbase = (unsigned)(uintptr_t)(void*)sw;  // flat addr low 32 bits == LDS offset
#pragma unroll
    for (int c = 0; c < 6; ++c) {
      unsigned loff = ldsbase + (unsigned)(c * 4096 + tid * 16);
      const void* g = (const char*)pk + c * 4096 + tid * 16;
      asm volatile("global_load_async_to_lds_b128 %0, %1, off"
                   :: "v"(loff), "v"(g) : "memory");
    }
    asm volatile("s_wait_asynccnt 0x0" ::: "memory");
  }

  int p  = (blockIdx.x * 8 + wave) * 16 + nloc;
  int pc = (p < npts) ? p : (npts - 1);   // clamp loads; WMMA needs full EXEC

  // ---- gather: build B1 (features, 32x16 f16, lane = point) ----
  const float* cp = coords + (size_t)pc * 3;
  float fx = (cp[0] + 1.f) * 0.5f;   // in [0,1] -> no lower clamp needed
  float fy = (cp[1] + 1.f) * 0.5f;
  float fz = (cp[2] + 1.f) * 0.5f;

  v16h B1v;
#pragma unroll
  for (int j = 0; j < 8; ++j) {
    // B layout: half h -> K = h + 16*hi; feature K -> level K>>1. So pair j -> level j + 8*hi.
    float    Rf   = hi ? (float)RES[j + 8] : (float)RES[j];
    unsigned Ru   = hi ? RES[j + 8] : RES[j];
    unsigned R2u  = hi ? RES[j + 8] * RES[j + 8] : RES[j] * RES[j];
    unsigned msk  = hi ? MASKV[j + 8] : MASKV[j];
    unsigned lofb = (hi ? (unsigned)(j + 8) : (unsigned)j) * (2u * MAX_TABLE * 4u); // byte off
    float Rm1 = Rf - 1.f;
    // cvt_u32_f32 truncates == floor for non-negative; upper clamp via one v_min
    unsigned dx = (unsigned)fminf(fx * Rf, Rm1);
    unsigned dy = (unsigned)fminf(fy * Rf, Rm1);
    unsigned dz = (unsigned)fminf(fz * Rf, Rm1);
    unsigned idx  = (dx * R2u + dy * Ru + dz) & msk;
    unsigned boff = lofb + idx * 8u;                     // 32-bit SADDR+voffset form
    float2 f = *(const float2*)((const char*)tables + boff);
    B1v[2 * j]     = (_Float16)f.x;
    B1v[2 * j + 1] = (_Float16)f.y;
  }

  // ---- out^T accumulators init with b2 (C-layout row M = 16*n2 + off + r) ----
  v8f acc[4];
#pragma unroll
  for (int n2 = 0; n2 < 4; ++n2) {
    const float* bp = b2 + n2 * 16 + off;
#pragma unroll
    for (int r = 0; r < 8; ++r) acc[n2][r] = bp[r];
  }

  __syncthreads();   // weights staged for all waves
  const v16h* __restrict__ swv = (const v16h*)sw;

  // ---- fused GEMM1 (h^T = W1^T x^T + b1) and GEMM2 (out^T += W2^T relu(h)) ----
#pragma unroll
  for (int kc = 0; kc < 4; ++kc) {
    // b1 as C operand for the two hidden tiles of this K-chunk
    v8f c0, c1;
    const float* bp0 = b1 + 32 * kc + off;
    const float* bp1 = b1 + 32 * kc + 16 + off;
#pragma unroll
    for (int r = 0; r < 8; ++r) { c0[r] = bp0[r]; c1[r] = bp1[r]; }

    v16h A0 = swv[(2 * kc + 0) * 32 + lane];
    v16h A1 = swv[(2 * kc + 1) * 32 + lane];
    v8f h0 = __builtin_amdgcn_wmma_f32_16x16x32_f16(false, A0, false, B1v, (short)0, c0, false, false);
    v8f h1 = __builtin_amdgcn_wmma_f32_16x16x32_f16(false, A1, false, B1v, (short)0, c1, false, false);

    // pack to f16 pairs (v_cvt_pk_f16_f32), ReLU in f16 (v_pk_max_num_f16),
    // exchange row-halves across lane^16 with ds_swizzle SWAPX16, select.
    v8h o0, o1;
#pragma unroll
    for (int r = 0; r < 8; ++r) {
      o0[r] = (_Float16)h0[r];
      o1[r] = (_Float16)h1[r];
    }
    v8h zz = {};
    o0 = __builtin_elementwise_max(o0, zz);
    o1 = __builtin_elementwise_max(o1, zz);

    union { v8h v; unsigned u[4]; } p0, p1;
    p0.v = o0; p1.v = o1;
    union { v16h v; unsigned u[8]; } B2;
#pragma unroll
    for (int d = 0; d < 4; ++d) {
      unsigned s0 = (unsigned)__builtin_amdgcn_ds_swizzle((int)p0.u[d], 0x401f); // SWAPX16
      unsigned s1 = (unsigned)__builtin_amdgcn_ds_swizzle((int)p1.u[d], 0x401f);
      B2.u[d]     = hi ? s1 : p0.u[d];
      B2.u[4 + d] = hi ? p1.u[d] : s0;
    }

#pragma unroll
    for (int n2 = 0; n2 < 4; ++n2) {
      v16h A2 = swv[(8 + n2 * 4 + kc) * 32 + lane];
      acc[n2] = __builtin_amdgcn_wmma_f32_16x16x32_f16(false, A2, false, B2.v, (short)0, acc[n2], false, false);
    }
  }

  // ---- store out[p][16*n2 + off + r] : contiguous 8-float runs per lane ----
  if (p < npts) {
    float* op = out + (size_t)p * 64 + off;
#pragma unroll
    for (int n2 = 0; n2 < 4; ++n2) {
#pragma unroll
      for (int r = 0; r < 8; ++r) op[n2 * 16 + r] = acc[n2][r];
    }
  }
}

extern "C" void kernel_launch(void* const* d_in, const int* in_sizes, int n_in,
                              void* d_out, int out_size, void* d_ws, size_t ws_size,
                              hipStream_t stream) {
  (void)n_in; (void)out_size; (void)ws_size;
  const float* coords = (const float*)d_in[0];
  const float* tables = (const float*)d_in[1];
  const float* W1     = (const float*)d_in[2];
  const float* b1     = (const float*)d_in[3];
  const float* W2     = (const float*)d_in[4];
  const float* b2     = (const float*)d_in[5];
  float* out = (float*)d_out;
  _Float16* pk = (_Float16*)d_ws;   // 24 KB weight prepack

  int npts = in_sizes[0] / 3;

  hg_prepack_kernel<<<3, 256, 0, stream>>>(W1, W2, pk);

  int nblk = (npts + 127) / 128;    // 8 waves/block * 16 points/wave
  hg_encode_mlp_kernel<<<nblk, 256, 0, stream>>>(coords, tables, b1, b2, pk, out, npts);
}
